// MultiHeadAttention_10170482556972
// MI455X (gfx1250) — compile-verified, matrix operand order checked by reading, not run on hardware
//
#include <hip/hip_runtime.h>
#include <hip/hip_bf16.h>

// ---------------------------------------------------------------------------
// MHA forward for MI455X (gfx1250, wave32, WMMA f32_16x16x32_f16).
// x (2,2048,1024) f32, W* (1024,1024) f32 -> out (2,2048,1024) f32.
// f16 matmuls, f32 accumulation; flash softmax; async double-buffered LDS.
// ---------------------------------------------------------------------------

typedef __attribute__((ext_vector_type(16))) _Float16 v16h;
typedef __attribute__((ext_vector_type(8)))  _Float16 v8h;
typedef __attribute__((ext_vector_type(4)))  _Float16 v4h;
typedef __attribute__((ext_vector_type(8)))  float    v8f;

#define DMODEL 1024
#define NHEADS 16
#define DK     64
#define SEQ    2048
#define BATCH  2
#define MROWS  (BATCH * SEQ)   // 4096
#define LDP    72              // LDS row stride (halves): 144B, bank-friendly

#ifndef __has_builtin
#define __has_builtin(x) 0
#endif

#if __has_builtin(__builtin_amdgcn_global_load_async_to_lds_b128)
#define HAVE_ASYNC 1
#else
#define HAVE_ASYNC 0
#endif

// Builtin parameter types (from hipcc diagnostic): vector_size(16) int* in
// AS1 (global source) and AS3 (LDS destination).
typedef int b128i __attribute__((vector_size(16)));
typedef __attribute__((address_space(1))) b128i b128i_glob;
typedef __attribute__((address_space(3))) b128i b128i_lds;

// 16B global -> LDS copy: async (ASYNCcnt) if available, else synchronous.
__device__ __forceinline__ void copy_b128(const _Float16* __restrict__ g,
                                          _Float16* l) {
#if HAVE_ASYNC
  __builtin_amdgcn_global_load_async_to_lds_b128(
      (b128i_glob*)g, (b128i_lds*)l, 0, 0);
#else
  *(v8h*)l = *(const v8h*)g;
#endif
}

#if HAVE_ASYNC
#if __has_builtin(__builtin_amdgcn_s_wait_asynccnt)
#define WAIT_ASYNC(N) __builtin_amdgcn_s_wait_asynccnt(N)
#else
#define WAIT_ASYNC(N) asm volatile("s_wait_asynccnt " #N ::: "memory")
#endif
#else
#define WAIT_ASYNC(N) do {} while (0)
#endif

// --- WMMA wrapper -----------------------------------------------------------
__device__ __forceinline__ v8f wmma_f16(v16h a, v16h b, v8f c) {
  return __builtin_amdgcn_wmma_f32_16x16x32_f16(
      false, a, false, b, (short)0, c, false, false);
}

// --- 16x32 f16 fragment load (A or contiguous-K B operand), stride `ld` ----
// ISA 7.12.2: lanes 0-15 -> K {0..7,16..23}, lanes 16-31 -> K {8..15,24..31}.
__device__ __forceinline__ v16h load_frag(const _Float16* __restrict__ base,
                                          int ld, int lane) {
  const int row  = lane & 15;
  const int koff = ((lane >> 4) & 1) * 8;
  const _Float16* p = base + row * ld + koff;
  v8h a = *(const v8h*)(p);
  v8h b = *(const v8h*)(p + 16);
  v16h r;
#pragma unroll
  for (int i = 0; i < 8; ++i) { r[i] = a[i]; r[i + 8] = b[i]; }
  return r;
}

// --- DPP row-of-16 max reduction (pure VALU, no DS traffic) -----------------
template <int CTRL>
__device__ __forceinline__ float dpp_max_step(float v) {
  int s = __builtin_amdgcn_mov_dpp(__builtin_bit_cast(int, v), CTRL, 0xF, 0xF, true);
  return fmaxf(v, __builtin_bit_cast(float, s));
}
__device__ __forceinline__ float rowmax16(float v) {
  v = dpp_max_step<0xB1>(v);    // quad_perm(1,0,3,2)  : xor 1
  v = dpp_max_step<0x4E>(v);    // quad_perm(2,3,0,1)  : xor 2
  v = dpp_max_step<0x141>(v);   // row_half_mirror     : xor 4 (groups uniform)
  v = dpp_max_step<0x140>(v);   // row_mirror          : xor 8 (groups uniform)
  return v;
}

// --- f32 -> f16 conversion, 4-wide -----------------------------------------
__global__ void cvt_f32_f16(const float* __restrict__ in,
                            _Float16* __restrict__ out, int n4) {
  int i = blockIdx.x * blockDim.x + threadIdx.x;
  if (i < n4) {
    float4 v = ((const float4*)in)[i];
    v4h o;
    o[0] = (_Float16)v.x; o[1] = (_Float16)v.y;
    o[2] = (_Float16)v.z; o[3] = (_Float16)v.w;
    ((v4h*)out)[i] = o;
  }
}

// --- GEMM: C(M,N) = A(M,K) * W^T, W stored (N,K), both f16 -----------------
// MODE 0: f16 -> (B,H,S,dk)  [Q/K projections; `oscale` folds 1/sqrt(dk) in Q]
// MODE 2: f16 -> (B,H,dk,S)  [V projection, transposed]
// MODE 4: f32 row-major       [final output projection]
// 128 threads = 4 waves; block tile 64x64; wave tile 32x32; 64-deep K slices
// of A and W staged in LDS, double-buffered via async loads.
template <int MODE>
__global__ __launch_bounds__(128)
void gemm16(const _Float16* __restrict__ A, const _Float16* __restrict__ W,
            void* __restrict__ out, int M, int N, int K, float oscale) {
  __shared__ _Float16 asbuf[2][64 * LDP];
  __shared__ _Float16 bsbuf[2][64 * LDP];

  const int tid  = threadIdx.x;
  const int lane = tid & 31;
  const int w    = tid >> 5;
  const int mblk = blockIdx.y * 64;
  const int nblk = blockIdx.x * 64;

  // stage one 64x64 tile of A and of W into LDS (8 x b128 per thread)
  auto stage = [&](int k0, int buf) {
#pragma unroll
    for (int c = 0; c < 4; ++c) {
      int chunk = tid * 4 + c;            // 0..511
      int row = chunk >> 3, part = chunk & 7;
      copy_b128(A + (size_t)(mblk + row) * K + k0 + part * 8,
                asbuf[buf] + row * LDP + part * 8);
    }
#pragma unroll
    for (int c = 0; c < 4; ++c) {
      int chunk = tid * 4 + c;
      int row = chunk >> 3, part = chunk & 7;
      copy_b128(W + (size_t)(nblk + row) * K + k0 + part * 8,
                bsbuf[buf] + row * LDP + part * 8);
    }
  };

  v8f acc[2][2] = {};
  const int lm = (w >> 1) * 32;           // wave's m offset in block tile
  const int ln = (w & 1) * 32;            // wave's n offset in block tile

  stage(0, 0);
  for (int k0 = 0; k0 < K; k0 += 64) {
    const int cur = (k0 >> 6) & 1;
    const bool more = (k0 + 64) < K;
    if (more) { stage(k0 + 64, cur ^ 1); WAIT_ASYNC(8); }
    else      { WAIT_ASYNC(0); }
    __syncthreads();

#pragma unroll
    for (int ks = 0; ks < 64; ks += 32) {
      v16h a0 = load_frag(asbuf[cur] + (lm +  0) * LDP + ks, LDP, lane);
      v16h a1 = load_frag(asbuf[cur] + (lm + 16) * LDP + ks, LDP, lane);
      v16h b0 = load_frag(bsbuf[cur] + (ln +  0) * LDP + ks, LDP, lane);
      v16h b1 = load_frag(bsbuf[cur] + (ln + 16) * LDP + ks, LDP, lane);
      acc[0][0] = wmma_f16(a0, b0, acc[0][0]);
      acc[0][1] = wmma_f16(a0, b1, acc[0][1]);
      acc[1][0] = wmma_f16(a1, b0, acc[1][0]);
      acc[1][1] = wmma_f16(a1, b1, acc[1][1]);
    }

    __syncthreads();                      // reads done before next overwrite
  }

  const int rbase = ((lane >> 4) & 1) << 3;
  const int col   = lane & 15;
#pragma unroll
  for (int i = 0; i < 2; ++i)
#pragma unroll
    for (int j = 0; j < 2; ++j)
#pragma unroll
      for (int r = 0; r < 8; ++r) {
        int m = mblk + lm + i * 16 + rbase + r;
        int n = nblk + ln + j * 16 + col;
        float v = acc[i][j][r] * oscale;
        if (MODE == 4) {
          ((float*)out)[(size_t)m * N + n] = v;
        } else {
          int b = m >> 11, s = m & (SEQ - 1);
          int h = n >> 6,  d = n & (DK - 1);
          size_t idx;
          if (MODE == 0)
            idx = (((size_t)(b * NHEADS + h)) * SEQ + s) * DK + d;
          else
            idx = (((size_t)(b * NHEADS + h)) * DK + d) * SEQ + s;
          ((_Float16*)out)[idx] = (_Float16)v;
        }
      }
}

// --- Flash attention: softmax(QK^T) V (scale pre-folded into Q) -------------
// Q,K: (B,H,S,dk) f16.  Vt: (B,H,dk,S) f16.  attn_out: (B,S,D) f16.
// Block = 4 waves x 16 q-rows.  64-key K/Vt tiles staged once per block into
// LDS (async, double-buffered), shared by all waves; softmax rescale and the
// barrier pair are amortized over 64 keys.
__global__ __launch_bounds__(128)
void attn_kernel(const _Float16* __restrict__ Q, const _Float16* __restrict__ Kd,
                 const _Float16* __restrict__ Vt, _Float16* __restrict__ attn_out) {
  __shared__ _Float16 kbuf[2][64 * LDP];  // 64 keys x 64 d
  __shared__ _Float16 vbuf[2][64 * LDP];  // 64 d x 64 keys
  __shared__ _Float16 pbuf[4][16 * LDP];  // per-wave P tile 16 x 64

  const int tid  = threadIdx.x;
  const int lane = tid & 31;
  const int w    = tid >> 5;
  const int bh   = blockIdx.y;
  const int b    = bh >> 4, h = bh & (NHEADS - 1);
  const int q0   = blockIdx.x * 64 + w * 16;

  const _Float16* Qh  = Q  + (size_t)bh * SEQ * DK;
  const _Float16* Kh  = Kd + (size_t)bh * SEQ * DK;
  const _Float16* Vth = Vt + (size_t)bh * DK * SEQ;
  _Float16* pb = pbuf[w];

  v16h qf0 = load_frag(Qh + (size_t)q0 * DK +  0, DK, lane);
  v16h qf1 = load_frag(Qh + (size_t)q0 * DK + 32, DK, lane);

  v16h ones;
#pragma unroll
  for (int i = 0; i < 16; ++i) ones[i] = (_Float16)1.0f;

  auto stage_kv = [&](int kb, int buf) {
#pragma unroll
    for (int c = 0; c < 4; ++c) {         // K tile: 64 rows x 8 parts
      int chunk = tid * 4 + c;            // 0..511
      int row = chunk >> 3, part = chunk & 7;
      copy_b128(Kh + (size_t)(kb + row) * DK + part * 8,
                kbuf[buf] + row * LDP + part * 8);
    }
#pragma unroll
    for (int c = 0; c < 4; ++c) {         // Vt tile: 64 rows x 8 parts
      int chunk = tid * 4 + c;
      int row = chunk >> 3, part = chunk & 7;
      copy_b128(Vth + (size_t)row * SEQ + kb + part * 8,
                vbuf[buf] + row * LDP + part * 8);
    }
  };

  const int rbase = ((lane >> 4) & 1) << 3;
  const int col   = lane & 15;

  float mr[8], lr[8];
#pragma unroll
  for (int r = 0; r < 8; ++r) { mr[r] = -1.0e30f; lr[r] = 0.0f; }
  v8f accO[4] = {};

  stage_kv(0, 0);
  for (int kb = 0; kb < SEQ; kb += 64) {
    const int cur = (kb >> 6) & 1;
    const bool more = (kb + 64) < SEQ;
    if (more) { stage_kv(kb + 64, cur ^ 1); WAIT_ASYNC(8); }
    else      { WAIT_ASYNC(0); }
    __syncthreads();

    // ---- scores: four 16x16 tiles; batch frag loads ahead of WMMAs ----
    v8f s0 = {}, s1 = {}, s2 = {}, s3 = {};
    {
      v16h ka = load_frag(kbuf[cur] +  0 * LDP +  0, LDP, lane);
      v16h kb0 = load_frag(kbuf[cur] +  0 * LDP + 32, LDP, lane);
      v16h kc = load_frag(kbuf[cur] + 16 * LDP +  0, LDP, lane);
      v16h kd = load_frag(kbuf[cur] + 16 * LDP + 32, LDP, lane);
      s0 = wmma_f16(qf0, ka, s0);  s0 = wmma_f16(qf1, kb0, s0);
      s1 = wmma_f16(qf0, kc, s1);  s1 = wmma_f16(qf1, kd, s1);
      ka = load_frag(kbuf[cur] + 32 * LDP +  0, LDP, lane);
      kb0 = load_frag(kbuf[cur] + 32 * LDP + 32, LDP, lane);
      kc = load_frag(kbuf[cur] + 48 * LDP +  0, LDP, lane);
      kd = load_frag(kbuf[cur] + 48 * LDP + 32, LDP, lane);
      s2 = wmma_f16(qf0, ka, s2);  s2 = wmma_f16(qf1, kb0, s2);
      s3 = wmma_f16(qf0, kc, s3);  s3 = wmma_f16(qf1, kd, s3);
    }

    // ---- online softmax over 64 keys: max via DPP, P -> LDS ----
#pragma unroll
    for (int r = 0; r < 8; ++r) {
      float a0 = s0[r], a1 = s1[r], a2 = s2[r], a3 = s3[r];
      float tmax = rowmax16(fmaxf(fmaxf(a0, a1), fmaxf(a2, a3)));
      float mnew = fmaxf(mr[r], tmax);
      float scl  = __expf(mr[r] - mnew);
      float p0 = __expf(a0 - mnew);
      float p1 = __expf(a1 - mnew);
      float p2 = __expf(a2 - mnew);
      float p3 = __expf(a3 - mnew);
      mr[r] = mnew;
      lr[r] *= scl;
#pragma unroll
      for (int t = 0; t < 4; ++t) accO[t][r] *= scl;
      int m = rbase + r;
      pb[m * LDP +      col] = (_Float16)p0;
      pb[m * LDP + 16 + col] = (_Float16)p1;
      pb[m * LDP + 32 + col] = (_Float16)p2;
      pb[m * LDP + 48 + col] = (_Float16)p3;
    }

    // ---- read P back as two A-fragments (4x ds_load_b128) ----
    v16h pf0, pf1;
    {
      const _Float16* pp = pb + (lane & 15) * LDP + ((lane >> 4) & 1) * 8;
      v8h a = *(const v8h*)(pp);
      v8h bq = *(const v8h*)(pp + 16);
      v8h c = *(const v8h*)(pp + 32);
      v8h d = *(const v8h*)(pp + 48);
#pragma unroll
      for (int i = 0; i < 8; ++i) {
        pf0[i] = a[i]; pf0[i + 8] = bq[i];
        pf1[i] = c[i]; pf1[i + 8] = d[i];
      }
    }

    // ---- row sums via matrix unit: P(16x64) x ones(64x16) ----
    v8f rs = {};
    rs = wmma_f16(pf0, ones, rs);
    rs = wmma_f16(pf1, ones, rs);
#pragma unroll
    for (int r = 0; r < 8; ++r) lr[r] += rs[r];

    // ---- O += P * V over both 32-key slices; batch frag loads ----
    {
      v16h vf0 = load_frag(vbuf[cur] +  0 * LDP + 0, LDP, lane);
      v16h vf1 = load_frag(vbuf[cur] + 16 * LDP + 0, LDP, lane);
      v16h vf2 = load_frag(vbuf[cur] + 32 * LDP + 0, LDP, lane);
      v16h vf3 = load_frag(vbuf[cur] + 48 * LDP + 0, LDP, lane);
      accO[0] = wmma_f16(pf0, vf0, accO[0]);
      accO[1] = wmma_f16(pf0, vf1, accO[1]);
      accO[2] = wmma_f16(pf0, vf2, accO[2]);
      accO[3] = wmma_f16(pf0, vf3, accO[3]);
      vf0 = load_frag(vbuf[cur] +  0 * LDP + 32, LDP, lane);
      vf1 = load_frag(vbuf[cur] + 16 * LDP + 32, LDP, lane);
      vf2 = load_frag(vbuf[cur] + 32 * LDP + 32, LDP, lane);
      vf3 = load_frag(vbuf[cur] + 48 * LDP + 32, LDP, lane);
      accO[0] = wmma_f16(pf1, vf0, accO[0]);
      accO[1] = wmma_f16(pf1, vf1, accO[1]);
      accO[2] = wmma_f16(pf1, vf2, accO[2]);
      accO[3] = wmma_f16(pf1, vf3, accO[3]);
    }

    __syncthreads();
  }

  // ---- normalize and scatter to (B,S,D) f16 ----
#pragma unroll
  for (int r = 0; r < 8; ++r) {
    float inv = 1.0f / lr[r];
    int s = q0 + rbase + r;
#pragma unroll
    for (int t = 0; t < 4; ++t) {
      attn_out[((size_t)b * SEQ + s) * DMODEL + h * DK + t * 16 + col] =
          (_Float16)(accO[t][r] * inv);
    }
  }
}

// ---------------------------------------------------------------------------
extern "C" void kernel_launch(void* const* d_in, const int* in_sizes, int n_in,
                              void* d_out, int out_size, void* d_ws, size_t ws_size,
                              hipStream_t stream) {
  const float* x  = (const float*)d_in[0];
  const float* Wq = (const float*)d_in[1];
  const float* Wk = (const float*)d_in[2];
  const float* Wv = (const float*)d_in[3];
  const float* Wo = (const float*)d_in[4];

  _Float16* ws = (_Float16*)d_ws;
  size_t off = 0;
  _Float16* x16  = ws + off; off += (size_t)MROWS * DMODEL;
  _Float16* Wq16 = ws + off; off += (size_t)DMODEL * DMODEL;
  _Float16* Wk16 = ws + off; off += (size_t)DMODEL * DMODEL;
  _Float16* Wv16 = ws + off; off += (size_t)DMODEL * DMODEL;
  _Float16* Wo16 = ws + off; off += (size_t)DMODEL * DMODEL;
  _Float16* Q16  = ws + off; off += (size_t)MROWS * DMODEL;   // (B,H,S,dk)
  _Float16* K16  = ws + off; off += (size_t)MROWS * DMODEL;   // (B,H,S,dk)
  _Float16* Vt16 = ws + off; off += (size_t)MROWS * DMODEL;   // (B,H,dk,S)
  _Float16* A16  = ws + off; off += (size_t)MROWS * DMODEL;   // (B,S,D)

  {
    int n4 = (MROWS * DMODEL) / 4;
    cvt_f32_f16<<<(n4 + 255) / 256, 256, 0, stream>>>(x, x16, n4);
    int w4 = (DMODEL * DMODEL) / 4;
    cvt_f32_f16<<<(w4 + 255) / 256, 256, 0, stream>>>(Wq, Wq16, w4);
    cvt_f32_f16<<<(w4 + 255) / 256, 256, 0, stream>>>(Wk, Wk16, w4);
    cvt_f32_f16<<<(w4 + 255) / 256, 256, 0, stream>>>(Wv, Wv16, w4);
    cvt_f32_f16<<<(w4 + 255) / 256, 256, 0, stream>>>(Wo, Wo16, w4);
  }

  dim3 gproj(DMODEL / 64, MROWS / 64);   // (16, 64)
  gemm16<0><<<gproj, 128, 0, stream>>>(x16, Wq16, (void*)Q16,  MROWS, DMODEL, DMODEL, 0.125f);
  gemm16<0><<<gproj, 128, 0, stream>>>(x16, Wk16, (void*)K16,  MROWS, DMODEL, DMODEL, 1.0f);
  gemm16<2><<<gproj, 128, 0, stream>>>(x16, Wv16, (void*)Vt16, MROWS, DMODEL, DMODEL, 1.0f);

  dim3 gattn(SEQ / 64, BATCH * NHEADS);  // (32, 32)
  attn_kernel<<<gattn, 128, 0, stream>>>(Q16, K16, Vt16, A16);

  gemm16<4><<<gproj, 128, 0, stream>>>(A16, Wo16, d_out, MROWS, DMODEL, DMODEL, 1.0f);
}
